// GPTLanguageModel_6459630813511
// MI455X (gfx1250) — matrix operand order repaired and checked by reading, not compiled, not run on hardware
//
#include <hip/hip_runtime.h>
#include <hip/hip_bf16.h>

// ---------------------------------------------------------------------------
// GPT forward on MI455X (gfx1250): wave32, WMMA f16 16x16x32, fp32 accum.
// f16 weight cache stored pre-transposed [N][K] (one-time convert+transpose),
// fused QKV GEMM, flash attention, async global->LDS tile copies (ASYNCcnt).
// ---------------------------------------------------------------------------

#define GPT_L  6
#define GPT_B  4
#define GPT_T  1024
#define GPT_D  768
#define GPT_H  12
#define GPT_HD 64
#define GPT_FF 3072
#define GPT_V  8192
#define GPT_M  (GPT_B * GPT_T)   // 4096 activation rows
#define QKV_S  (3 * GPT_D)       // 2304: fused qkv row stride

typedef __attribute__((ext_vector_type(16))) _Float16 v16h;
typedef __attribute__((ext_vector_type(8)))  _Float16 v8h;
typedef __attribute__((ext_vector_type(4)))  _Float16 v4h;
typedef __attribute__((ext_vector_type(8)))  float    v8f;
typedef __attribute__((ext_vector_type(4)))  int      v4i;

#if defined(__has_builtin)
# if __has_builtin(__builtin_amdgcn_global_load_async_to_lds_b128)
#  define USE_ASYNC_LDS 1
# endif
#endif
#ifndef USE_ASYNC_LDS
# define USE_ASYNC_LDS 0
#endif

#if USE_ASYNC_LDS
typedef __attribute__((address_space(1))) v4i gv4i;   // global int4
typedef __attribute__((address_space(3))) v4i lv4i;   // LDS int4
__device__ __forceinline__ void async_cp16(const _Float16* g, _Float16* l) {
  __builtin_amdgcn_global_load_async_to_lds_b128((gv4i*)g, (lv4i*)l, 0, 0);
}
# if __has_builtin(__builtin_amdgcn_s_wait_asynccnt)
#  define ASYNC_WAIT0() __builtin_amdgcn_s_wait_asynccnt(0)
# else
#  define ASYNC_WAIT0() asm volatile("s_wait_asynccnt 0x0" ::: "memory")
# endif
#endif

// Load a 16-half WMMA fragment as two contiguous 16B LDS reads.
__device__ __forceinline__ v16h ld_frag(const _Float16* p0, const _Float16* p1) {
  v8h lo = *(const v8h*)p0;
  v8h hi = *(const v8h*)p1;
  return __builtin_shufflevector(lo, hi, 0,1,2,3,4,5,6,7,8,9,10,11,12,13,14,15);
}

// ---------------------------------------------------------------------------
// Tiled transpose + fp32->f16: dst[n*K + k] = (f16)src[k*N + n], per layer z.
// ---------------------------------------------------------------------------
__global__ __launch_bounds__(256)
void transpose_cvt_kernel(const float* __restrict__ src, _Float16* __restrict__ dst,
                          int K, int N, size_t srcLS, size_t dstLS)
{
  __shared__ _Float16 t[32][33];
  const float* s = src + (size_t)blockIdx.z * srcLS;
  _Float16*    d = dst + (size_t)blockIdx.z * dstLS;
  const int bk = blockIdx.y * 32;
  const int bn = blockIdx.x * 32;
  const int r  = threadIdx.x >> 3;        // 0..31
  const int c4 = (threadIdx.x & 7) * 4;   // 0..28
  const float4 f = *(const float4*)&s[(size_t)(bk + r) * N + bn + c4];
  t[r][c4 + 0] = (_Float16)f.x;
  t[r][c4 + 1] = (_Float16)f.y;
  t[r][c4 + 2] = (_Float16)f.z;
  t[r][c4 + 3] = (_Float16)f.w;
  __syncthreads();
  v4h o;
  o[0] = t[c4 + 0][r]; o[1] = t[c4 + 1][r];
  o[2] = t[c4 + 2][r]; o[3] = t[c4 + 3][r];
  *(v4h*)&d[(size_t)(bn + r) * K + bk + c4] = o;
}

// ---------------------------------------------------------------------------
// GEMM: C[M,N] = act(A_f16[M,K] * Wt_f16[N,K]^T + bias [+ residual])
// 256 threads = 8 waves, tile 128x128, K-step 32, double-buffered LDS,
// 2-stage pipeline, one barrier per K-step. A and Wt tiles are loaded with
// identical contiguous-16B patterns (async-to-LDS when available).
// Wave grid 2(M) x 4(N); each wave: 4x2 tiles -> 8 wmma per K-step.
// ---------------------------------------------------------------------------
#define GBM 128
#define GBN 128
#define GBK 32
#define GLD 40   // 32 + 8 halves pad (80B rows -> 16B-aligned chunks)

template<bool BIAS, bool RESID, bool RELU, bool OUTF, bool OUTH>
__global__ __launch_bounds__(256)
void gemm_f16_kernel(const _Float16* __restrict__ A, const _Float16* __restrict__ Wt,
                     const float* __restrict__ bias, const float* __restrict__ resid,
                     float* __restrict__ Cf, _Float16* __restrict__ Ch,
                     int M, int N, int K)
{
  __shared__ _Float16 As[2][GBM * GLD];   // [row][k]
  __shared__ _Float16 Bt[2][GBN * GLD];   // [col][k]

  const int tid  = threadIdx.x;
  const int lane = tid & 31;
  const int wave = tid >> 5;
  const int hi16 = lane >> 4;
  const int lm   = lane & 15;
  const int bm = blockIdx.y * GBM;
  const int bn = blockIdx.x * GBN;
  const int wm = (wave >> 2) * 64;
  const int wn = (wave & 3) * 32;

  // chunk coords: 128 rows x 4 x 16B chunks = 512 chunks, 2 per thread
  int rr[2], cc[2];
  #pragma unroll
  for (int i = 0; i < 2; ++i) {
    int c = tid + i * 256;
    rr[i] = c >> 2;  cc[i] = (c & 3) * 8;
  }

  const int ntiles = K / GBK;

#if USE_ASYNC_LDS
  #pragma unroll
  for (int i = 0; i < 2; ++i) {
    async_cp16(&A [(size_t)(bm + rr[i]) * K + cc[i]], &As[0][rr[i] * GLD + cc[i]]);
    async_cp16(&Wt[(size_t)(bn + rr[i]) * K + cc[i]], &Bt[0][rr[i] * GLD + cc[i]]);
  }
  ASYNC_WAIT0();
#else
  v8h ar[2], br[2];
  #pragma unroll
  for (int i = 0; i < 2; ++i) {
    ar[i] = *(const v8h*)&A [(size_t)(bm + rr[i]) * K + cc[i]];
    br[i] = *(const v8h*)&Wt[(size_t)(bn + rr[i]) * K + cc[i]];
  }
  #pragma unroll
  for (int i = 0; i < 2; ++i) {
    *(v8h*)&As[0][rr[i] * GLD + cc[i]] = ar[i];
    *(v8h*)&Bt[0][rr[i] * GLD + cc[i]] = br[i];
  }
#endif
  __syncthreads();

  v8f acc[4][2] = {};

  for (int t = 0; t < ntiles; ++t) {
    const int cur = t & 1, nxt = cur ^ 1;
    const bool pf = (t + 1 < ntiles);
    if (pf) {
      const int k1 = (t + 1) * GBK;
#if USE_ASYNC_LDS
      #pragma unroll
      for (int i = 0; i < 2; ++i) {
        async_cp16(&A [(size_t)(bm + rr[i]) * K + k1 + cc[i]], &As[nxt][rr[i] * GLD + cc[i]]);
        async_cp16(&Wt[(size_t)(bn + rr[i]) * K + k1 + cc[i]], &Bt[nxt][rr[i] * GLD + cc[i]]);
      }
#else
      #pragma unroll
      for (int i = 0; i < 2; ++i) {
        ar[i] = *(const v8h*)&A [(size_t)(bm + rr[i]) * K + k1 + cc[i]];
        br[i] = *(const v8h*)&Wt[(size_t)(bn + rr[i]) * K + k1 + cc[i]];
      }
#endif
      if (t + 2 < ntiles)   // stream ahead (global_prefetch_b8)
        __builtin_prefetch(&Wt[(size_t)(bn + rr[0]) * K + (t + 2) * GBK + cc[0]], 0, 0);
    }

    // ---- fragments + 8 WMMA from LDS[cur] ---------------------------------
    v16h af[4], bf[2];
    #pragma unroll
    for (int mi = 0; mi < 4; ++mi) {
      const _Float16* p = &As[cur][(wm + mi * 16 + lm) * GLD];
      af[mi] = ld_frag(p + hi16 * 8, p + 16 + hi16 * 8);
    }
    #pragma unroll
    for (int nj = 0; nj < 2; ++nj) {
      const _Float16* p = &Bt[cur][(wn + nj * 16 + lm) * GLD];
      bf[nj] = ld_frag(p + hi16 * 8, p + 16 + hi16 * 8);
    }
    #pragma unroll
    for (int mi = 0; mi < 4; ++mi)
      #pragma unroll
      for (int nj = 0; nj < 2; ++nj)
        acc[mi][nj] = __builtin_amdgcn_wmma_f32_16x16x32_f16(
            false, af[mi], false, bf[nj], (short)0, acc[mi][nj], false, false);

    if (pf) {
#if USE_ASYNC_LDS
      ASYNC_WAIT0();
#else
      #pragma unroll
      for (int i = 0; i < 2; ++i) {
        *(v8h*)&As[nxt][rr[i] * GLD + cc[i]] = ar[i];
        *(v8h*)&Bt[nxt][rr[i] * GLD + cc[i]] = br[i];
      }
#endif
    }
    __syncthreads();
  }

  // ---- epilogue (fully specialized, branch-free) --------------------------
  #pragma unroll
  for (int mi = 0; mi < 4; ++mi)
    #pragma unroll
    for (int nj = 0; nj < 2; ++nj) {
      const int col = bn + wn + nj * 16 + lm;
      const float bsv = BIAS ? bias[col] : 0.0f;
      #pragma unroll
      for (int vr = 0; vr < 8; ++vr) {
        const int row = bm + wm + mi * 16 + vr + hi16 * 8;
        float v = acc[mi][nj][vr] + bsv;
        if (RESID) v += resid[(size_t)row * N + col];
        if (RELU)  v = fmaxf(v, 0.0f);
        if (OUTF)  Cf[(size_t)row * N + col] = v;
        if (OUTH)  Ch[(size_t)row * N + col] = (_Float16)v;
      }
    }
}

// ---------------------------------------------------------------------------
// Flash attention over the fused qkv buffer (row stride QKV_S).
// One WG = 128 query rows of one (b,h). 8 waves x 16 q-rows.
// ---------------------------------------------------------------------------
#define FTQ 128
#define FTK 64
#define FLD 72   // 64 + 8 halves pad

__global__ __launch_bounds__(256)
void flash_attn_kernel(const _Float16* __restrict__ QKV, _Float16* __restrict__ O)
{
  __shared__ _Float16 Qs[FTQ * FLD];
  __shared__ _Float16 Ks[FTK * FLD];   // [key][d]
  __shared__ _Float16 Vt[FTK * FLD];   // [d][key]
  __shared__ _Float16 Ps[FTQ * FLD];   // per-wave P staging (16 rows each)

  const int tid  = threadIdx.x;
  const int lane = tid & 31;
  const int wave = tid >> 5;
  const int hi16 = lane >> 4;
  const int lm   = lane & 15;
  const int qb   = blockIdx.x;
  const int b    = blockIdx.y / GPT_H;
  const int h    = blockIdx.y % GPT_H;
  const size_t qkvbase = (size_t)b * GPT_T * QKV_S + (size_t)h * GPT_HD;
  const _Float16* Q  = QKV + qkvbase;                 // cols [0,D)
  const _Float16* Kb = QKV + qkvbase + GPT_D;         // cols [D,2D)
  const _Float16* Vb = QKV + qkvbase + 2 * GPT_D;     // cols [2D,3D)
  const size_t obase = (size_t)b * GPT_T * GPT_D + (size_t)h * GPT_HD;

  #pragma unroll
  for (int i = 0; i < 4; ++i) {
    int c = tid + i * 256;
    int r = c >> 3, co = (c & 7) * 8;
    *(v8h*)&Qs[r * FLD + co] =
        *(const v8h*)&Q[(size_t)(qb * FTQ + r) * QKV_S + co];
  }
  __syncthreads();

  v16h qa[2];
  {
    const _Float16* p = &Qs[(wave * 16 + lm) * FLD];
    qa[0] = ld_frag(p + hi16 * 8,      p + 16 + hi16 * 8);
    qa[1] = ld_frag(p + 32 + hi16 * 8, p + 48 + hi16 * 8);
  }

  v8f  o[4] = {};
  float mrow[8], lrow[8];
  #pragma unroll
  for (int vr = 0; vr < 8; ++vr) { mrow[vr] = -3.0e38f; lrow[vr] = 0.0f; }

  const int nkt = 2 * qb + 2;
  for (int kt = 0; kt < nkt; ++kt) {
    __syncthreads();   // K/V buffer reuse
    #pragma unroll
    for (int i = 0; i < 2; ++i) {
      int c = tid + i * 256;
      int r = c >> 3, co = (c & 7) * 8;
      *(v8h*)&Ks[r * FLD + co] =
          *(const v8h*)&Kb[(size_t)(kt * FTK + r) * QKV_S + co];
      v8h vv = *(const v8h*)&Vb[(size_t)(kt * FTK + r) * QKV_S + co];
      #pragma unroll
      for (int j = 0; j < 8; ++j) Vt[(co + j) * FLD + r] = vv[j];
    }
    if (kt + 1 < nkt)
      __builtin_prefetch(&Kb[(size_t)((kt + 1) * FTK + (tid >> 3)) * QKV_S], 0, 0);
    __syncthreads();

    // S = Q K^T  (HD=64 -> 2 chained WMMA per 16x16 tile)
    v8f s[4] = {};
    #pragma unroll
    for (int nj = 0; nj < 4; ++nj) {
      const _Float16* p = &Ks[(nj * 16 + lm) * FLD];
      v16h b0 = ld_frag(p + hi16 * 8,      p + 16 + hi16 * 8);
      v16h b1 = ld_frag(p + 32 + hi16 * 8, p + 48 + hi16 * 8);
      s[nj] = __builtin_amdgcn_wmma_f32_16x16x32_f16(false, qa[0], false, b0,
                                                     (short)0, s[nj], false, false);
      s[nj] = __builtin_amdgcn_wmma_f32_16x16x32_f16(false, qa[1], false, b1,
                                                     (short)0, s[nj], false, false);
    }

    // online causal softmax (per C-layout row vr)
    const int qrow0 = qb * FTQ + wave * 16 + hi16 * 8;
    const int kcol  = kt * FTK + lm;
    #pragma unroll
    for (int vr = 0; vr < 8; ++vr) {
      const int qrow = qrow0 + vr;
      float sv[4], rmax = -3.0e38f;
      #pragma unroll
      for (int nj = 0; nj < 4; ++nj) {
        float x = s[nj][vr] * 0.125f;                // HD^-0.5
        if (kcol + nj * 16 > qrow) x = -3.0e38f;
        sv[nj] = x;
        rmax = fmaxf(rmax, x);
      }
      #pragma unroll
      for (int off = 1; off < 16; off <<= 1)
        rmax = fmaxf(rmax, __shfl_xor(rmax, off, 32));
      const float mnew = fmaxf(mrow[vr], rmax);
      const float corr = __expf(mrow[vr] - mnew);
      float psum = 0.0f;
      #pragma unroll
      for (int nj = 0; nj < 4; ++nj) {
        float pv = __expf(sv[nj] - mnew);
        psum += pv;
        Ps[(wave * 16 + hi16 * 8 + vr) * FLD + nj * 16 + lm] = (_Float16)pv;
      }
      #pragma unroll
      for (int off = 1; off < 16; off <<= 1)
        psum += __shfl_xor(psum, off, 32);
      lrow[vr] = lrow[vr] * corr + psum;
      mrow[vr] = mnew;
      #pragma unroll
      for (int nj = 0; nj < 4; ++nj) o[nj][vr] *= corr;
    }

    // O += P V  (per-wave LDS region; same-wave DScnt ordering, no barrier)
    const _Float16* pp = &Ps[(wave * 16 + lm) * FLD];
    v16h pa0 = ld_frag(pp + hi16 * 8,      pp + 16 + hi16 * 8);
    v16h pa1 = ld_frag(pp + 32 + hi16 * 8, pp + 48 + hi16 * 8);
    #pragma unroll
    for (int nj = 0; nj < 4; ++nj) {
      const _Float16* pv = &Vt[(nj * 16 + lm) * FLD];
      v16h b0 = ld_frag(pv + hi16 * 8,      pv + 16 + hi16 * 8);
      v16h b1 = ld_frag(pv + 32 + hi16 * 8, pv + 48 + hi16 * 8);
      o[nj] = __builtin_amdgcn_wmma_f32_16x16x32_f16(false, pa0, false, b0,
                                                     (short)0, o[nj], false, false);
      o[nj] = __builtin_amdgcn_wmma_f32_16x16x32_f16(false, pa1, false, b1,
                                                     (short)0, o[nj], false, false);
    }
  }

  #pragma unroll
  for (int nj = 0; nj < 4; ++nj)
    #pragma unroll
    for (int vr = 0; vr < 8; ++vr) {
      const int qrow = qb * FTQ + wave * 16 + hi16 * 8 + vr;
      O[obase + (size_t)qrow * GPT_D + nj * 16 + lm] =
          (_Float16)(o[nj][vr] / lrow[vr]);
    }
}

// ---------------------------------------------------------------------------
// LayerNorm: fp32 in -> f16 out. One block per row.
// ---------------------------------------------------------------------------
__global__ __launch_bounds__(256)
void layernorm_kernel(const float* __restrict__ X, const float* __restrict__ g,
                      const float* __restrict__ b, _Float16* __restrict__ Y, int Dd)
{
  __shared__ float red[256];
  const int row = blockIdx.x, tid = threadIdx.x;
  const float* x = X + (size_t)row * Dd;
  float s1 = 0.0f, s2 = 0.0f;
  for (int i = tid; i < Dd; i += 256) { float v = x[i]; s1 += v; s2 += v * v; }
  red[tid] = s1; __syncthreads();
  for (int off = 128; off > 0; off >>= 1) { if (tid < off) red[tid] += red[tid + off]; __syncthreads(); }
  const float mean = red[0] / Dd; __syncthreads();
  red[tid] = s2; __syncthreads();
  for (int off = 128; off > 0; off >>= 1) { if (tid < off) red[tid] += red[tid + off]; __syncthreads(); }
  const float var  = red[0] / Dd - mean * mean;
  const float rstd = rsqrtf(var + 1e-5f);
  for (int i = tid; i < Dd; i += 256)
    Y[(size_t)row * Dd + i] = (_Float16)((x[i] - mean) * rstd * g[i] + b[i]);
}

// ---------------------------------------------------------------------------
// Embedding
// ---------------------------------------------------------------------------
__global__ void embed_kernel(const int* __restrict__ idx, const float* __restrict__ tok,
                             const float* __restrict__ pos, float* __restrict__ X)
{
  const int bt = blockIdx.x;
  const int t  = bt % GPT_T;
  const int tk = idx[bt];
  const float* te = tok + (size_t)tk * GPT_D;
  const float* pe = pos + (size_t)t  * GPT_D;
  float* x = X + (size_t)bt * GPT_D;
  for (int i = threadIdx.x; i < GPT_D; i += blockDim.x) x[i] = te[i] + pe[i];
}

// ---------------------------------------------------------------------------
// Loss
// ---------------------------------------------------------------------------
__global__ void zero_loss_kernel(float* loss) { *loss = 0.0f; }

__global__ __launch_bounds__(256)
void loss_kernel(const float* __restrict__ logits, const int* __restrict__ targets,
                 float* __restrict__ loss, float invN)
{
  __shared__ float red[256];
  const int row = blockIdx.x, tid = threadIdx.x;
  const float* lr = logits + (size_t)row * GPT_V;
  float mx = -3.0e38f;
  for (int i = tid; i < GPT_V; i += 256) mx = fmaxf(mx, lr[i]);
  red[tid] = mx; __syncthreads();
  for (int off = 128; off > 0; off >>= 1) { if (tid < off) red[tid] = fmaxf(red[tid], red[tid + off]); __syncthreads(); }
  mx = red[0]; __syncthreads();
  float se = 0.0f;
  for (int i = tid; i < GPT_V; i += 256) se += __expf(lr[i] - mx);
  red[tid] = se; __syncthreads();
  for (int off = 128; off > 0; off >>= 1) { if (tid < off) red[tid] += red[tid + off]; __syncthreads(); }
  if (tid == 0) {
    const float lse = mx + __logf(red[0]);
    atomicAdd(loss, -(lr[targets[row]] - lse) * invN);
  }
}

// ---------------------------------------------------------------------------
// Host-side launch
// ---------------------------------------------------------------------------
static inline size_t align256(size_t x) { return (x + 255) & ~(size_t)255; }

extern "C" void kernel_launch(void* const* d_in, const int* in_sizes, int n_in,
                              void* d_out, int out_size, void* d_ws, size_t ws_size,
                              hipStream_t stream) {
  const int*   idx     = (const int*)  d_in[0];
  const int*   targets = (const int*)  d_in[1];
  const float* tok_emb = (const float*)d_in[2];
  const float* pos_emb = (const float*)d_in[3];
  const float* Wq = (const float*)d_in[4];
  const float* Wk = (const float*)d_in[5];
  const float* Wv = (const float*)d_in[6];
  const float* Wp = (const float*)d_in[7];
  const float* bp = (const float*)d_in[8];
  const float* W1 = (const float*)d_in[9];
  const float* b1 = (const float*)d_in[10];
  const float* W2 = (const float*)d_in[11];
  const float* b2 = (const float*)d_in[12];
  const float* g1 = (const float*)d_in[13];
  const float* be1= (const float*)d_in[14];
  const float* g2 = (const float*)d_in[15];
  const float* be2= (const float*)d_in[16];
  const float* gf = (const float*)d_in[17];
  const float* bf = (const float*)d_in[18];
  const float* Wh = (const float*)d_in[19];
  const float* bh = (const float*)d_in[20];

  float* out  = (float*)d_out;                      // [B*T*V logits][1 loss]
  float* loss = out + (size_t)GPT_M * GPT_V;

  // ---- workspace carve ----------------------------------------------------
  char* ws = (char*)d_ws;
  size_t off = 0;
  float*    x     = (float*)   (ws + off); off = align256(off + (size_t)GPT_M * GPT_D * 4);
  _Float16* h16   = (_Float16*)(ws + off); off = align256(off + (size_t)GPT_M * GPT_D * 2);
  _Float16* qkv16 = (_Float16*)(ws + off); off = align256(off + (size_t)GPT_M * QKV_S * 2);
  _Float16* a16   = (_Float16*)(ws + off); off = align256(off + (size_t)GPT_M * GPT_D * 2);
  _Float16* f16   = (_Float16*)(ws + off); off = align256(off + (size_t)GPT_M * GPT_FF * 2);
  // f16 transposed weight caches [N][K] (~100MB, L2-resident on reuse)
  const size_t DD = (size_t)GPT_D * GPT_D;
  const size_t DF = (size_t)GPT_D * GPT_FF;
  const size_t DV = (size_t)GPT_D * GPT_V;
  _Float16* wqkvT = (_Float16*)(ws + off); off = align256(off + (size_t)GPT_L * 3 * DD * 2);
  _Float16* wpT   = (_Float16*)(ws + off); off = align256(off + (size_t)GPT_L * DD * 2);
  _Float16* w1T   = (_Float16*)(ws + off); off = align256(off + (size_t)GPT_L * DF * 2);
  _Float16* w2T   = (_Float16*)(ws + off); off = align256(off + (size_t)GPT_L * DF * 2);
  _Float16* whT   = (_Float16*)(ws + off); off = align256(off + DV * 2);
  (void)off; (void)n_in; (void)in_sizes; (void)out_size; (void)ws_size;

  const dim3 blk(256);
  const dim3 gQKV(QKV_S  / GBN, GPT_M / GBM);   // N=2304
  const dim3 gD  (GPT_D  / GBN, GPT_M / GBM);
  const dim3 gFF (GPT_FF / GBN, GPT_M / GBM);
  const dim3 gV  (GPT_V  / GBN, GPT_M / GBM);
  const dim3 gFA (GPT_T / FTQ, GPT_B * GPT_H);

  // ---- one-time weight transpose + f16 conversion -------------------------
  auto tr = [&](const float* s, _Float16* d, int K, int N, size_t sLS, size_t dLS, int nl) {
    dim3 g(N / 32, K / 32, nl);
    transpose_cvt_kernel<<<g, blk, 0, stream>>>(s, d, K, N, sLS, dLS);
  };
  tr(Wq, wqkvT,          GPT_D, GPT_D, DD, 3 * DD, GPT_L);   // rows [0,D)
  tr(Wk, wqkvT + DD,     GPT_D, GPT_D, DD, 3 * DD, GPT_L);   // rows [D,2D)
  tr(Wv, wqkvT + 2 * DD, GPT_D, GPT_D, DD, 3 * DD, GPT_L);   // rows [2D,3D)
  tr(Wp, wpT, GPT_D,  GPT_D,  DD, DD, GPT_L);
  tr(W1, w1T, GPT_D,  GPT_FF, DF, DF, GPT_L);
  tr(W2, w2T, GPT_FF, GPT_D,  DF, DF, GPT_L);
  tr(Wh, whT, GPT_D,  GPT_V,  DV, DV, 1);

  embed_kernel<<<GPT_M, blk, 0, stream>>>(idx, tok_emb, pos_emb, x);

  for (int l = 0; l < GPT_L; ++l) {
    const size_t oD = (size_t)l * GPT_D;
    const size_t oF = (size_t)l * GPT_FF;

    layernorm_kernel<<<GPT_M, blk, 0, stream>>>(x, g1 + oD, be1 + oD, h16, GPT_D);
    gemm_f16_kernel<false,false,false,false,true><<<gQKV, blk, 0, stream>>>(
        h16, wqkvT + (size_t)l * 3 * DD, nullptr, nullptr, nullptr, qkv16,
        GPT_M, QKV_S, GPT_D);
    flash_attn_kernel<<<gFA, blk, 0, stream>>>(qkv16, a16);
    gemm_f16_kernel<true,true,false,true,false><<<gD, blk, 0, stream>>>(
        a16, wpT + (size_t)l * DD, bp + oD, x, x, nullptr, GPT_M, GPT_D, GPT_D);
    layernorm_kernel<<<GPT_M, blk, 0, stream>>>(x, g2 + oD, be2 + oD, h16, GPT_D);
    gemm_f16_kernel<true,false,true,false,true><<<gFF, blk, 0, stream>>>(
        h16, w1T + (size_t)l * DF, b1 + oF, nullptr, nullptr, f16, GPT_M, GPT_FF, GPT_D);
    gemm_f16_kernel<true,true,false,true,false><<<gD, blk, 0, stream>>>(
        f16, w2T + (size_t)l * DF, b2 + oD, x, x, nullptr, GPT_M, GPT_D, GPT_FF);
  }

  layernorm_kernel<<<GPT_M, blk, 0, stream>>>(x, gf, bf, h16, GPT_D);
  gemm_f16_kernel<true,false,false,true,false><<<gV, blk, 0, stream>>>(
      h16, whT, bh, nullptr, out, nullptr, GPT_M, GPT_V, GPT_D);

  zero_loss_kernel<<<1, 1, 0, stream>>>(loss);
  loss_kernel<<<GPT_M, blk, 0, stream>>>(out, targets, loss, 1.0f / (float)GPT_M);
}
